// BitLinear_56281251447179
// MI455X (gfx1250) — matrix operand order repaired and checked by reading, not compiled, not run on hardware
//
#include <hip/hip_runtime.h>

typedef __attribute__((ext_vector_type(16))) _Float16 v16h;
typedef __attribute__((ext_vector_type(4)))  _Float16 v4h;
typedef __attribute__((ext_vector_type(8)))  float    v8f;
typedef __attribute__((ext_vector_type(4)))  unsigned int u32x4;
typedef __attribute__((ext_vector_type(2)))  unsigned int u32x2;
typedef __attribute__((ext_vector_type(8)))  int i32x8;
typedef __attribute__((ext_vector_type(4)))  int i32x4;

#define D_K   4096        // inner (d) dimension
#define D_N   4096        // output (o) dimension
#define MROWS 8192        // B*S
#define BM    128
#define BN    128
#define BK    32
#define LDA   40          // half-stride of ldsA rows (80B: conflict-free, 16B aligned)
#define LDB   132         // half-stride of ldsB rows (264B: conflict-free, 8B aligned)

// dynamic-LDS byte offsets (compile-time constants for the TDM D#)
#define OFF_A0 0
#define OFF_A1 (BM * LDA * 2)                    // 10240
#define OFF_B0 (2 * BM * LDA * 2)                // 20480
#define OFF_B1 (2 * BM * LDA * 2 + BK * LDB * 2) // 28928
#define SMEM_BYTES (2 * BM * LDA * 2 + 2 * BK * LDB * 2) // 37376

union AFrag { v16h h; u32x4 q[2]; };
union BFrag { v16h h; u32x2 q[4]; };

static __device__ inline u32x2 mk2(unsigned a, unsigned b) { u32x2 r; r[0]=a; r[1]=b; return r; }

#if defined(__HIP_DEVICE_COMPILE__) && __has_builtin(__builtin_amdgcn_tensor_load_to_lds)
#define USE_TDM 1
#endif

#ifdef USE_TDM
struct TdmDesc { u32x4 g0; i32x8 g1; };

// Build a 2D TDM tile descriptor: tile_d0 contiguous f16 x tile_d1 rows,
// row stride = stride0 elements; LDS rows padded pad_amount(+1) DW per interval.
// Built in UNIFORM code so it stays in SGPRs.
static __device__ __forceinline__ TdmDesc tdm_make_2d(
    unsigned lds_off, unsigned long long ga,
    unsigned tensor_d0, unsigned tensor_d1, unsigned stride0,
    unsigned tile_d0, unsigned tile_d1,
    unsigned pad_interval, unsigned pad_amount)
{
    TdmDesc d;
    d.g0[0] = 1u;                                          // count=1
    d.g0[1] = lds_off;                                     // lds_addr
    d.g0[2] = (unsigned)(ga & 0xFFFFFFFFu);                // global_addr lo
    d.g0[3] = (unsigned)((ga >> 32) & 0x01FFFFFFu) | (2u << 30); // hi | type=2
    d.g1[0] = (int)((1u << 16)                 // data_size = 2 bytes
                  | (1u << 20)                 // pad_enable
                  | (pad_interval << 22)
                  | (pad_amount << 25));
    d.g1[1] = (int)((tensor_d0 & 0xFFFFu) << 16);
    d.g1[2] = (int)((tensor_d0 >> 16) | ((tensor_d1 & 0xFFFFu) << 16));
    d.g1[3] = (int)((tensor_d1 >> 16) | (tile_d0 << 16));
    d.g1[4] = (int)(tile_d1 & 0xFFFFu);
    d.g1[5] = (int)stride0;                                // dim0_stride lo32
    d.g1[6] = 0;
    d.g1[7] = 0;
    return d;
}

static __device__ __forceinline__ void tdm_issue(TdmDesc d) {
    i32x4 gz; gz[0]=0; gz[1]=0; gz[2]=0; gz[3]=0;
#if __clang_major__ >= 23
    i32x8 gz8; gz8[0]=0; gz8[1]=0; gz8[2]=0; gz8[3]=0; gz8[4]=0; gz8[5]=0; gz8[6]=0; gz8[7]=0;
    __builtin_amdgcn_tensor_load_to_lds(d.g0, d.g1, gz, gz, gz8, 0);
#else
    __builtin_amdgcn_tensor_load_to_lds(d.g0, d.g1, gz, gz, 0);
#endif
}
#endif

// ---------------- kernel 1: partial sums of |W| ----------------
__global__ __launch_bounds__(256) void bl_abssum(const float* __restrict__ W,
                                                 float* __restrict__ partial) {
    __shared__ float red[256];
    const size_t n4 = (size_t)D_N * D_K / 4;
    float s = 0.0f;
    for (size_t i = (size_t)blockIdx.x * 256 + threadIdx.x; i < n4;
         i += (size_t)gridDim.x * 256) {
        float4 v = ((const float4*)W)[i];
        s += fabsf(v.x) + fabsf(v.y) + fabsf(v.z) + fabsf(v.w);
    }
    red[threadIdx.x] = s;
    __syncthreads();
    for (int off = 128; off > 0; off >>= 1) {
        if ((int)threadIdx.x < off) red[threadIdx.x] += red[threadIdx.x + off];
        __syncthreads();
    }
    if (threadIdx.x == 0) partial[blockIdx.x] = red[0];
}

// ---------------- kernel 2: finalize gamma ----------------
__global__ __launch_bounds__(256) void bl_finalize(const float* __restrict__ partial,
                                                   float* __restrict__ gbuf) {
    __shared__ float red[256];
    float s = partial[threadIdx.x] + partial[threadIdx.x + 256];
    red[threadIdx.x] = s;
    __syncthreads();
    for (int off = 128; off > 0; off >>= 1) {
        if ((int)threadIdx.x < off) red[threadIdx.x] += red[threadIdx.x + off];
        __syncthreads();
    }
    if (threadIdx.x == 0) {
        float gamma = red[0] * (1.0f / ((float)D_N * (float)D_K));
        gbuf[0] = gamma;                          // output scale
        gbuf[1] = (gamma == 0.0f) ? 1.0f : gamma; // safe divisor
    }
}

// ---------------- kernel 3: ternary quantize W and transpose -> WqT[k][n] (f16) ----------------
__global__ __launch_bounds__(256) void bl_quantT(const float* __restrict__ W,
                                                 _Float16* __restrict__ WqT,
                                                 const float* __restrict__ gbuf) {
    __shared__ _Float16 tile[32][33];
    const float inv = 1.0f / gbuf[1];
    const int n0 = blockIdx.x * 32;
    const int k0 = blockIdx.y * 32;
    const int tx = threadIdx.x;   // 0..31
    const int ty = threadIdx.y;   // 0..7
#pragma unroll
    for (int i = 0; i < 4; ++i) {
        float w = W[(size_t)(n0 + ty + i * 8) * D_K + k0 + tx];
        float q = rintf(w * inv);
        q = fminf(1.0f, fmaxf(-1.0f, q));
        tile[ty + i * 8][tx] = (_Float16)q;
    }
    __syncthreads();
#pragma unroll
    for (int i = 0; i < 4; ++i)
        WqT[(size_t)(k0 + ty + i * 8) * D_N + n0 + tx] = tile[tx][ty + i * 8];
}

// ---------------- kernel 4: X f32 -> f16 ----------------
__global__ __launch_bounds__(256) void bl_cvt(const float* __restrict__ X,
                                              _Float16* __restrict__ Xh) {
    size_t i = (size_t)blockIdx.x * 256 + threadIdx.x;   // one float4 per thread
    float4 v = ((const float4*)X)[i];
    v4h o = { (_Float16)v.x, (_Float16)v.y, (_Float16)v.z, (_Float16)v.w };
    ((v4h*)Xh)[i] = o;
}

// ---------------- kernel 5: WMMA GEMM  Out[m][n] = gamma * sum_k Xh[m][k]*WqT[k][n] ----------------
__global__ __launch_bounds__(256) void bl_gemm(const _Float16* __restrict__ Xh,
                                               const _Float16* __restrict__ WqT,
                                               const float* __restrict__ gbuf,
                                               float* __restrict__ Out) {
    extern __shared__ _Float16 smem[];
    _Float16* ldsA0 = smem + OFF_A0 / 2;
    _Float16* ldsA1 = smem + OFF_A1 / 2;
    _Float16* ldsB0 = smem + OFF_B0 / 2;
    _Float16* ldsB1 = smem + OFF_B1 / 2;

    const int t    = threadIdx.x;
    const int lane = t & 31;
    const int wave = t >> 5;       // 8 waves
    const int wm   = wave & 3;     // M sub-block: 32 rows each
    const int wn   = wave >> 2;    // N sub-block: 64 cols each
    const int gm0  = blockIdx.y * BM;
    const int gn0  = blockIdx.x * BN;

    v8f acc[2][4];
#pragma unroll
    for (int i = 0; i < 2; ++i)
#pragma unroll
        for (int j = 0; j < 4; ++j)
#pragma unroll
            for (int r = 0; r < 8; ++r) acc[i][j][r] = 0.0f;

#ifdef USE_TDM
    // -------- TDM staging: uniform descriptors, one wave issues --------
    unsigned long long gaA = (unsigned long long)(uintptr_t)(Xh + (size_t)gm0 * D_K);
    unsigned long long gaB = (unsigned long long)(uintptr_t)(WqT + gn0);
    {
        // A: 32 (k contig) x 128 (m rows), pad 4 DW / 16 DW -> 80B rows
        TdmDesc dA = tdm_make_2d(OFF_A0, gaA, D_K, MROWS, D_K, BK, BM, 3, 3);
        // B: 128 (n contig) x 32 (k rows), pad 2 DW / 64 DW -> 264B rows
        TdmDesc dB = tdm_make_2d(OFF_B0, gaB, D_N, D_K, D_N, BN, BK, 5, 1);
        if (wave == 0) { tdm_issue(dA); tdm_issue(dB); }
    }
    gaA += (unsigned long long)BK * 2;           // advance K by 32 halves
    gaB += (unsigned long long)BK * D_N * 2;     // advance 32 rows
#else
    const int arow  = t >> 1;            // 0..127
    const int acolh = (t & 1) * 16;      // halves
    const int brow  = t >> 3;            // 0..31 (k)
    const int bcolh = (t & 7) * 16;      // halves (n)
    const _Float16* gA = Xh  + (size_t)(gm0 + arow) * D_K + acolh;
    const _Float16* gB = WqT + (size_t)brow * D_N + gn0 + bcolh;
    {
        u32x4 ra0 = ((const u32x4*)gA)[0];
        u32x4 ra1 = ((const u32x4*)gA)[1];
        u32x4 rb0 = ((const u32x4*)gB)[0];
        u32x4 rb1 = ((const u32x4*)gB)[1];
        u32x4* dA = (u32x4*)(ldsA0 + arow * LDA + acolh);
        dA[0] = ra0; dA[1] = ra1;
        u32x2* dB = (u32x2*)(ldsB0 + brow * LDB + bcolh);
        dB[0] = mk2(rb0[0], rb0[1]); dB[1] = mk2(rb0[2], rb0[3]);
        dB[2] = mk2(rb1[0], rb1[1]); dB[3] = mk2(rb1[2], rb1[3]);
    }
#endif

    const int mrow  = lane & 15;
    const int khalf = lane >> 4;
    int buf = 0;

    for (int k0 = 0; k0 < D_K; k0 += BK) {
        const bool hasNext = (k0 + BK) < D_K;

#ifdef USE_TDM
        // SOPP, EXEC-independent; waves 1-7 have TENSORcnt==0 so this is free
        __builtin_amdgcn_s_wait_tensorcnt(0);
        __syncthreads();   // publish DMA'd buffer; fences last iter's reads
        if (hasNext) {
            TdmDesc dA = tdm_make_2d(buf ? OFF_A0 : OFF_A1, gaA, D_K, MROWS, D_K, BK, BM, 3, 3);
            TdmDesc dB = tdm_make_2d(buf ? OFF_B0 : OFF_B1, gaB, D_N, D_K, D_N, BN, BK, 5, 1);
            if (wave == 0) { tdm_issue(dA); tdm_issue(dB); }
            gaA += (unsigned long long)BK * 2;
            gaB += (unsigned long long)BK * D_N * 2;
        }
#else
        u32x4 ra0, ra1, rb0, rb1;
        if (hasNext) {
            const _Float16* nA = gA + (k0 + BK);
            const _Float16* nB = gB + (size_t)(k0 + BK) * D_N;
            ra0 = ((const u32x4*)nA)[0];
            ra1 = ((const u32x4*)nA)[1];
            rb0 = ((const u32x4*)nB)[0];
            rb1 = ((const u32x4*)nB)[1];
        }
#endif
        const _Float16* curA = buf ? ldsA1 : ldsA0;
        const _Float16* curB = buf ? ldsB1 : ldsB0;

        // A fragments: 16-bit A 16x32 layout; two aligned quads per lane -> ds_load_b128
        AFrag a[2];
#pragma unroll
        for (int tm = 0; tm < 2; ++tm) {
            const u32x4* rowq = (const u32x4*)(curA + (wm * 32 + tm * 16 + mrow) * LDA);
            a[tm].q[0] = rowq[khalf];
            a[tm].q[1] = rowq[2 + khalf];
        }
        // B fragments: lane L holds column n=L%16, k = 8*(L/16)+s (s<8), 16+8*(L/16)+(s-8) (s>=8)
        BFrag b[4];
#pragma unroll
        for (int tn = 0; tn < 4; ++tn) {
            const _Float16* base = curB + khalf * 8 * LDB + wn * 64 + tn * 16 + mrow;
#pragma unroll
            for (int s = 0; s < 8; ++s) {
                b[tn].h[s]     = base[s * LDB];
                b[tn].h[8 + s] = base[(16 + s) * LDB];
            }
        }
#pragma unroll
        for (int tm = 0; tm < 2; ++tm)
#pragma unroll
            for (int tn = 0; tn < 4; ++tn)
                acc[tm][tn] = __builtin_amdgcn_wmma_f32_16x16x32_f16(
                    false, a[tm].h, false, b[tn].h, (short)0, acc[tm][tn],
                    false, false);

#ifdef USE_TDM
        if (hasNext) buf ^= 1;
#else
        if (hasNext) {
            _Float16* nxtA = buf ? ldsA0 : ldsA1;
            _Float16* nxtB = buf ? ldsB0 : ldsB1;
            u32x4* dA = (u32x4*)(nxtA + arow * LDA + acolh);
            dA[0] = ra0; dA[1] = ra1;
            u32x2* dB = (u32x2*)(nxtB + brow * LDB + bcolh);
            dB[0] = mk2(rb0[0], rb0[1]); dB[1] = mk2(rb0[2], rb0[3]);
            dB[2] = mk2(rb1[0], rb1[1]); dB[3] = mk2(rb1[2], rb1[3]);
            __syncthreads();
            buf ^= 1;
        }
#endif
    }

    // epilogue: C/D layout — VGPR r: lanes 0-15 -> M=r, lanes 16-31 -> M=r+8; N=lane%16
    const float gamma = gbuf[0];
    const int nlo = lane & 15;
    const int mhi = (lane >> 4) * 8;
#pragma unroll
    for (int tm = 0; tm < 2; ++tm)
#pragma unroll
        for (int tn = 0; tn < 4; ++tn) {
            const int mbase = gm0 + wm * 32 + tm * 16 + mhi;
            const int n     = gn0 + wn * 64 + tn * 16 + nlo;
#pragma unroll
            for (int r = 0; r < 8; ++r)
                Out[(size_t)(mbase + r) * D_N + n] = acc[tm][tn][r] * gamma;
        }
}

extern "C" void kernel_launch(void* const* d_in, const int* in_sizes, int n_in,
                              void* d_out, int out_size, void* d_ws, size_t ws_size,
                              hipStream_t stream) {
    const float* x = (const float*)d_in[0];   // (4,2048,4096) f32
    const float* w = (const float*)d_in[1];   // (4096,4096) f32
    float* out = (float*)d_out;               // (4,2048,4096) f32

    // workspace layout
    float* gbuf    = (float*)d_ws;            // [0]=gamma, [1]=safe gamma
    float* partial = gbuf + 16;               // 512 partial sums
    char*  base    = (char*)d_ws + 4096;
    _Float16* xh   = (_Float16*)base;                                   // 64 MiB
    _Float16* wqT  = (_Float16*)(base + (size_t)MROWS * D_K * 2);       // 32 MiB

    bl_abssum  <<<512, 256, 0, stream>>>(w, partial);
    bl_finalize<<<1,   256, 0, stream>>>(partial, gbuf);
    bl_quantT  <<<dim3(D_N / 32, D_K / 32), dim3(32, 8), 0, stream>>>(w, wqT, gbuf);
    bl_cvt     <<<(int)((size_t)MROWS * D_K / 4 / 256), 256, 0, stream>>>(x, xh);
    bl_gemm    <<<dim3(D_N / BN, MROWS / BM), 256, SMEM_BYTES, stream>>>(xh, wqT, gbuf, out);
}
